// SpatialVectorQuantizer_76089640616043
// MI455X (gfx1250) — matrix operand order repaired and checked by reading, not compiled, not run on hardware
//
#include <hip/hip_runtime.h>
#include <math.h>

typedef __bf16 bf16;
typedef __attribute__((ext_vector_type(16))) __bf16 v16bf;
typedef __attribute__((ext_vector_type(8)))  float  v8f;
typedef __attribute__((ext_vector_type(4)))  unsigned int v4u;
typedef __attribute__((ext_vector_type(4)))  int    v4i;
typedef __attribute__((ext_vector_type(8)))  int    v8i;

#define NUM_EMB 1024
#define EMB_DIM 64
#define HW      1024          // 32*32
#define NROWS   32768         // 32 * HW
#define NELEM   2097152       // 32*64*32*32

// ---- workspace layout (float offsets) ----
#define WS_WNORM 0            // [1024] f32   ||w_n||^2
#define WS_WBF   1024         // bf16[1024*64] = 32768 floats of space
#define WS_CNT   33792        // [1024] f32   counts
#define WS_DW    34816        // [1024*64] f32 segment-sum of flat
#define WS_IDX   100352       // int[32768]   argmin indices
#define WS_SCAL  133120       // [0]=loss_sum [1]=n_total [2]=entropy_sum

// ---- d_out layout (reference tuple order, flattened) ----
#define O_Q      0
#define O_LOSS   2097152
#define O_IDX    2097153
#define O_PERP   2129921
#define O_ECS    2129922
#define O_EMAW   2130946
#define O_W      2196482

#define LDS_ROW   144                    // 128 B bf16 row + 16 B pad
#define LDS_WN    (NUM_EMB * LDS_ROW)    // 147456: wnorm f32[1024]
#define LDS_BYTES (LDS_WN + NUM_EMB * 4) // 151552

static __device__ __forceinline__ bf16 f2bf(float x) {
    union { float f; unsigned u; } v; v.f = x;
    unsigned r = v.u + 0x7FFFu + ((v.u >> 16) & 1u);   // round-to-nearest-even
    unsigned short h = (unsigned short)(r >> 16);
    return __builtin_bit_cast(bf16, h);
}

// TDM: DMA bf16 codebook [1024 rows x 64 elems x 2B] from global to LDS,
// padding each 128B row to 144B (pad_interval = 32 DWORDs, pad_amount = 4 DWORDs).
static __device__ __forceinline__ void tdm_load_codebook(const void* gaddr) {
    unsigned long long ga = (unsigned long long)gaddr;
    v4u g0;
    g0[0] = 1u;                                        // count=1 (valid), user mode
    g0[1] = 0u;                                        // lds_addr = 0
    g0[2] = (unsigned)(ga & 0xFFFFFFFFu);              // global_addr[31:0]
    g0[3] = (unsigned)((ga >> 32) & 0x01FFFFFFu)       // global_addr[56:32]
          | 0x80000000u;                               // type = 2 ("image")
    v8i g1;
    g1[0] = (1 << 16)          // data_size = 2 bytes
          | (1 << 20)          // pad_enable
          | (4 << 22)          // pad_interval: 32 DWORDs (128 B)
          | (3 << 25);         // pad_amount: 4 DWORDs (16 B)
    g1[1] = (EMB_DIM & 0xFFFF) << 16;                  // tensor_dim0 = 64
    g1[2] = (NUM_EMB & 0xFFFF) << 16;                  // tensor_dim1 = 1024
    g1[3] = (EMB_DIM & 0xFFFF) << 16;                  // tile_dim0   = 64
    g1[4] = (NUM_EMB & 0xFFFF);                        // tile_dim1   = 1024
    g1[5] = EMB_DIM;                                   // tensor_dim0_stride = 64
    g1[6] = 0;
    g1[7] = 0;
    v4i z4 = {0, 0, 0, 0};
#if defined(__clang_major__) && (__clang_major__ >= 23)
    v8i z8 = {0, 0, 0, 0, 0, 0, 0, 0};
    __builtin_amdgcn_tensor_load_to_lds(g0, g1, z4, z4, z8, 0);
#else
    __builtin_amdgcn_tensor_load_to_lds(g0, g1, z4, z4, 0);
#endif
}

// ---------------- kernel 0: init ws (zero accum, wnorm, bf16 codebook) ------
__global__ __launch_bounds__(256) void vq_init_kernel(
    const float* __restrict__ weight, float* __restrict__ ws)
{
    int t = blockIdx.x * 256 + threadIdx.x;          // grid 256 -> t in [0,65536)
    ws[WS_DW + t] = 0.0f;
    bf16* wbf = (bf16*)(ws + WS_WBF);
    wbf[t] = f2bf(weight[t]);
    if (t < NUM_EMB) {
        float s = 0.0f;
        #pragma unroll 8
        for (int d = 0; d < EMB_DIM; ++d) {
            float w = weight[t * EMB_DIM + d];
            s += w * w;
        }
        ws[WS_WNORM + t] = s;
        ws[WS_CNT + t] = 0.0f;
    }
    if (t < 3) ws[WS_SCAL + t] = 0.0f;
}

// ------------- kernel 1: WMMA bf16 distance GEMM + fused argmin ------------
// block = 256 (8 waves); each wave owns 32 rows (two 16-row A tiles), block
// owns 256 rows; grid 128 covers M = 32768. N loop over 1024 codes in
// 16-wide tiles; K = 64 -> 4x v_wmma_f32_16x16x32_bf16 per tile pair,
// with one shared B operand (4 ds_load_b128) per iteration.
__global__ __launch_bounds__(256) void vq_argmin_kernel(
    const float* __restrict__ inputs, float* __restrict__ ws)
{
    extern __shared__ char lds[];
    const int tid  = threadIdx.x;
    const int wave = tid >> 5, lane = tid & 31;
    const int half = lane >> 4, lr = lane & 15;

    // wave 0 kicks the Tensor Data Mover for the codebook; others proceed to
    // wnorm staging and meet at the barrier.
    if (tid < 32) {
        tdm_load_codebook(ws + WS_WBF);
    }
    for (int i = tid; i < NUM_EMB; i += 256)
        *(float*)(lds + LDS_WN + i * 4) = ws[WS_WNORM + i];
    if (tid < 32) {
        __builtin_amdgcn_s_wait_tensorcnt(0);
    }
    __syncthreads();

    const int b   = blockIdx.x >> 2;
    const int hw0 = (blockIdx.x & 3) << 8;                // 256-row tile, one image
    const float* xin = inputs + (size_t)b * (EMB_DIM * HW);
    const int row0 = hw0 + wave * 32 + lr;                // A tile 0 row (M)
    const int row1 = row0 + 16;                           // A tile 1 row (M)

    // A layout: lanes 0-15 M rows w/ K=0..15, lanes 16-31 K=16..31;
    // element e -> K = 16*half + e (matches B contiguous order).
    v16bf a0, a1, a2, a3;
    #pragma unroll
    for (int e = 0; e < 16; ++e) {
        int K = e + 16 * half;
        a0[e] = f2bf(xin[K * HW + row0]);                 // K in [0,32)
        a1[e] = f2bf(xin[(K + 32) * HW + row0]);          // K in [32,64)
        a2[e] = f2bf(xin[K * HW + row1]);
        a3[e] = f2bf(xin[(K + 32) * HW + row1]);
    }

    float best[16]; int bidx[16];
    #pragma unroll
    for (int r = 0; r < 16; ++r) { best[r] = 3.4e38f; bidx[r] = 0; }

    for (int n0 = 0; n0 < NUM_EMB; n0 += 16) {
        // B operand: lane -> N = n0+lr; lanes 0-15 hold K=0..15, lanes 16-31
        // hold K=16..31 (contiguous order, same as A) -> contiguous 32 B per lane.
        const char* bp = lds + (n0 + lr) * LDS_ROW + half * 32;
        union { uint4 u[2]; v16bf v; } ub0, ub1;
        ub0.u[0] = *(const uint4*)(bp);
        ub0.u[1] = *(const uint4*)(bp + 16);
        ub1.u[0] = *(const uint4*)(bp + 64);              // K += 32
        ub1.u[1] = *(const uint4*)(bp + 80);
        float wn = *(const float*)(lds + LDS_WN + (n0 + lr) * 4);

        v8f c0 = {0.f,0.f,0.f,0.f,0.f,0.f,0.f,0.f};
        v8f c1 = {0.f,0.f,0.f,0.f,0.f,0.f,0.f,0.f};
        c0 = __builtin_amdgcn_wmma_f32_16x16x32_bf16(
                 false, a0, false, ub0.v, (short)0, c0, false, false);
        c1 = __builtin_amdgcn_wmma_f32_16x16x32_bf16(
                 false, a2, false, ub0.v, (short)0, c1, false, false);
        c0 = __builtin_amdgcn_wmma_f32_16x16x32_bf16(
                 false, a1, false, ub1.v, (short)0, c0, false, false);
        c1 = __builtin_amdgcn_wmma_f32_16x16x32_bf16(
                 false, a3, false, ub1.v, (short)0, c1, false, false);

        // dist(m, n) = ||w||^2 - 2 x.w   (||x||^2 constant per row)
        #pragma unroll
        for (int r = 0; r < 8; ++r) {                     // C vgpr r -> M = r+8*half
            float d0 = wn - 2.0f * c0[r];
            if (d0 < best[r])     { best[r]     = d0; bidx[r]     = n0 + lr; }
            float d1 = wn - 2.0f * c1[r];
            if (d1 < best[8 + r]) { best[8 + r] = d1; bidx[8 + r] = n0 + lr; }
        }
    }

    // reduce argmin across the 16 lanes holding the same M row (tie -> lower idx)
    int* idxout = (int*)(ws + WS_IDX);
    #pragma unroll
    for (int r = 0; r < 16; ++r) {
        float d = best[r]; int i = bidx[r];
        for (int m = 1; m <= 8; m <<= 1) {
            float od = __shfl_xor(d, m, 32);
            int   oi = __shfl_xor(i, m, 32);
            if (od < d || (od == d && oi < i)) { d = od; i = oi; }
        }
        if (lr == 0) {
            int row = hw0 + wave * 32 + (r >> 3) * 16 + (r & 7) + 8 * half;
            idxout[b * HW + row] = i;
        }
    }
}

// -------- kernel 2: gather quantized, loss, counts/dw segment sums ---------
__global__ __launch_bounds__(64) void vq_gather_kernel(
    const float* __restrict__ inputs, const float* __restrict__ weight,
    float* __restrict__ ws, float* __restrict__ out)
{
    int m = blockIdx.x * 64 + threadIdx.x;                // grid 512
    const int* idxs = (const int*)(ws + WS_IDX);
    int idx = idxs[m];
    int b = m >> 10, hw = m & 1023;
    const float* x = inputs + (size_t)b * (EMB_DIM * HW) + hw;
    float*       q = out + O_Q + (size_t)b * (EMB_DIM * HW) + hw;
    float*      dw = ws + WS_DW + idx * EMB_DIM;

    float ls = 0.0f;
    #pragma unroll 4
    for (int d = 0; d < EMB_DIM; ++d) {
        float w  = weight[idx * EMB_DIM + d];
        float xv = x[d * HW];
        q[d * HW] = w;                                    // quantized_st == quantized
        float df = w - xv;
        ls += df * df;
        atomicAdd(&dw[d], xv);                            // dw = enc.T @ flat
    }
    out[O_IDX + m] = (float)idx;
    atomicAdd(&ws[WS_SCAL + 0], ls);
    atomicAdd(&ws[WS_CNT + idx], 1.0f);
}

// -------- kernel 3a: EMA cluster size, n_total, entropy --------------------
__global__ __launch_bounds__(256) void vq_ema_stats_kernel(
    const float* __restrict__ ema_cs, float* __restrict__ ws,
    float* __restrict__ out)
{
    int n = blockIdx.x * 256 + threadIdx.x;               // grid 4
    float c    = ws[WS_CNT + n];
    float necs = 0.99f * ema_cs[n] + 0.01f * c;
    out[O_ECS + n] = necs;
    atomicAdd(&ws[WS_SCAL + 1], necs);                    // n_total
    float p = c * (1.0f / (float)NROWS);
    atomicAdd(&ws[WS_SCAL + 2], p * logf(p + 1e-10f));    // sum p*log(p+eps)
}

// -------- kernel 3b: new ema_w, new weight, scalars ------------------------
__global__ __launch_bounds__(256) void vq_finalize_kernel(
    const float* __restrict__ ema_w, float* __restrict__ ws,
    float* __restrict__ out)
{
    int t = blockIdx.x * 256 + threadIdx.x;               // grid 256 -> 65536
    float ntot = ws[WS_SCAL + 1];
    int n = t >> 6;
    float necs = out[O_ECS + n];
    float cs = (necs + 1e-5f) / (ntot + NUM_EMB * 1e-5f) * ntot;
    float nema = 0.99f * ema_w[t] + 0.01f * ws[WS_DW + t];
    out[O_EMAW + t] = nema;
    out[O_W + t]    = nema / cs;
    if (t == 0) {
        out[O_LOSS] = 0.25f * ws[WS_SCAL + 0] * (1.0f / (float)NELEM);
        out[O_PERP] = expf(-ws[WS_SCAL + 2]);
    }
}

extern "C" void kernel_launch(void* const* d_in, const int* in_sizes, int n_in,
                              void* d_out, int out_size, void* d_ws, size_t ws_size,
                              hipStream_t stream)
{
    const float* inputs = (const float*)d_in[0];   // [32,64,32,32]
    const float* weight = (const float*)d_in[1];   // [1024,64]
    const float* ema_cs = (const float*)d_in[2];   // [1024]
    const float* ema_w  = (const float*)d_in[3];   // [1024,64]
    float* out = (float*)d_out;
    float* ws  = (float*)d_ws;

    vq_init_kernel  <<<256, 256, 0, stream>>>(weight, ws);
    vq_argmin_kernel<<<128, 256, LDS_BYTES, stream>>>(inputs, ws);
    vq_gather_kernel<<<512, 64,  0, stream>>>(inputs, weight, ws, out);
    vq_ema_stats_kernel<<<4, 256, 0, stream>>>(ema_cs, ws, out);
    vq_finalize_kernel <<<256, 256, 0, stream>>>(ema_w, ws, out);
}
